// SparseAttention_72395968741609
// MI455X (gfx1250) — matrix operand order repaired and checked by reading, not compiled, not run on hardware
//
#include <hip/hip_runtime.h>
#include <hip/hip_bf16.h>

// Problem constants (match reference)
#define B_    2
#define H_    16
#define HKV_  4
#define S_    4096
#define D_    64
#define BN_   64
#define G_    64
#define NB_   64
#define BPQ_  6
#define SCALE_ 0.125f      // 1/sqrt(64)
#define NEG_  -1e30f
#define KS_   72           // padded row stride (elements) for K ([key][d]) in LDS
                           // 72*2B = 144B: multiple of 16B (aligned b128 loads),
                           // 36-dword row step; a wave's K stores fill one row -> no conflict on K stores

typedef __attribute__((ext_vector_type(16))) __bf16 v16bf;
typedef __attribute__((ext_vector_type(8)))  __bf16 v8bf;
typedef __attribute__((ext_vector_type(2)))  __bf16 v2bf;
typedef __attribute__((ext_vector_type(8)))  float  v8f;

// Build a 16-wide bf16 fragment from two contiguous 16-byte LDS loads.
__device__ __forceinline__ v16bf load_frag16(const __bf16* p) {
    v8bf lo = *(const v8bf*)p;
    v8bf hi = *(const v8bf*)(p + 8);
    return __builtin_shufflevector(lo, hi, 0,1,2,3,4,5,6,7,8,9,10,11,12,13,14,15);
}

__launch_bounds__(128)
__global__ void sparse_attn_wmma(const float* __restrict__ q,
                                 const float* __restrict__ k,
                                 const float* __restrict__ v,
                                 const int*   __restrict__ seg,
                                 const int*   __restrict__ bidx,
                                 const float* __restrict__ slopes,
                                 float* __restrict__ out)
{
    __shared__ __align__(16) __bf16 sKt[D_ * KS_];     // K: [key][d], padded rows
    __shared__ __align__(16) __bf16 sV [BN_ * D_];     // V transposed: [d][key]
    __shared__ __align__(16) __bf16 sP [4][16 * BN_];  // per-wave P: [row][key]

    const int tid  = threadIdx.x;
    const int wv   = tid >> 5;               // wave id 0..3 -> M-tile
    const int lane = tid & 31;
    const int half = lane >> 4;              // 0: lanes 0-15, 1: lanes 16-31
    const int lx   = lane & 15;

    int id = blockIdx.x;
    const int qb = id % NB_;  id /= NB_;
    const int h  = id % H_;   id /= H_;
    const int b  = id;
    const int hkv = h / (H_ / HKV_);

    const int   qbase = qb * BN_ + wv * 16;
    const float slope = slopes[h];

    // ---- Q fragments in A layout (two contiguous 8-float runs per chunk) ----
    const float* qp = q + (((size_t)b * H_ + h) * S_ + (qbase + lx)) * D_;
    v16bf aq0, aq1;
    {
        v8f a0lo = *(const v8f*)(qp + 8 * half);
        v8f a0hi = *(const v8f*)(qp + 16 + 8 * half);
        v8f a1lo = *(const v8f*)(qp + 32 + 8 * half);
        v8f a1hi = *(const v8f*)(qp + 48 + 8 * half);
        #pragma unroll
        for (int j = 0; j < 8; ++j) {
            aq0[j]     = (__bf16)(SCALE_ * a0lo[j]);
            aq0[j + 8] = (__bf16)(SCALE_ * a0hi[j]);
            aq1[j]     = (__bf16)(SCALE_ * a1lo[j]);
            aq1[j + 8] = (__bf16)(SCALE_ * a1hi[j]);
        }
    }

    // per-lane metadata for the 8 C-layout rows this lane owns (row = r + 8*half)
    float qposr[8]; int qsegr[8];
    #pragma unroll
    for (int r = 0; r < 8; ++r) {
        const int m = qbase + r + 8 * half;
        qposr[r] = (float)m;
        qsegr[r] = seg[(size_t)b * S_ + m];
    }

    v8f accO[4];
    float mrun[8], lrun[8];
    #pragma unroll
    for (int r = 0; r < 8; ++r) {
        mrun[r] = NEG_;
        lrun[r] = 0.0f;
        #pragma unroll
        for (int nt = 0; nt < 4; ++nt) accO[nt][r] = 0.0f;
    }

    const bool denseQ = (qb == 0);           // query block 0 attends all keys
    const int  nkb    = denseQ ? NB_ : (1 + BPQ_);

    const float* kb_p = k + ((size_t)b * HKV_ + hkv) * (size_t)S_ * D_;
    const float* vb_p = v + ((size_t)b * HKV_ + hkv) * (size_t)S_ * D_;

    for (int s = 0; s < nkb; ++s) {
        const int kb = denseQ ? s : (s == 0 ? 0 : bidx[qb * BPQ_ + (s - 1)]);
        if (kb < 0) continue;                // uniform across the workgroup
        const int kpos0 = kb * BN_;

        __syncthreads();
        {   // cooperative stage: K as [key][d] (padded rows), V transposed as [d][key]
            const float2* ks2 = (const float2*)(kb_p + (size_t)kpos0 * D_);
            const float2* vs2 = (const float2*)(vb_p + (size_t)kpos0 * D_);
            for (int i = tid; i < BN_ * D_ / 2; i += 128) {
                const int key = i >> 5;          // 32 float2 per key row
                const int dp  = i & 31;
                float2 kk2 = ks2[i];             // coalesced global read
                sKt[key * KS_ + 2 * dp]     = (__bf16)kk2.x;
                sKt[key * KS_ + 2 * dp + 1] = (__bf16)kk2.y;
                float2 vv2 = vs2[i];
                sV[(2 * dp)     * BN_ + key] = (__bf16)vv2.x;
                sV[(2 * dp + 1) * BN_ + key] = (__bf16)vv2.y;
            }
        }
        __syncthreads();

        // ---- S = Q K^T : B-frag = (key=lx, depth in two 8-element runs like A) ----
        float sc[4][8];
        #pragma unroll
        for (int nt = 0; nt < 4; ++nt) {
            const __bf16* krow = &sKt[(nt * 16 + lx) * KS_];
            v16bf bk0 = __builtin_shufflevector(*(const v8bf*)(krow + 8 * half), *(const v8bf*)(krow + 16 + 8 * half), 0,1,2,3,4,5,6,7,8,9,10,11,12,13,14,15);
            v16bf bk1 = __builtin_shufflevector(*(const v8bf*)(krow + 32 + 8 * half), *(const v8bf*)(krow + 48 + 8 * half), 0,1,2,3,4,5,6,7,8,9,10,11,12,13,14,15);
            v8f c;
            #pragma unroll
            for (int r = 0; r < 8; ++r) c[r] = 0.0f;
            c = __builtin_amdgcn_wmma_f32_16x16x32_bf16(false, aq0, false, bk0, (short)0, c, false, false);
            c = __builtin_amdgcn_wmma_f32_16x16x32_bf16(false, aq1, false, bk1, (short)0, c, false, false);

            const int   kpos  = kpos0 + nt * 16 + lx;
            const int   kse   = seg[(size_t)b * S_ + kpos];
            const float kposf = (float)kpos;
            #pragma unroll
            for (int r = 0; r < 8; ++r) {
                const float bias = -slope * fabsf(qposr[r] - kposf);
                sc[nt][r] = (qsegr[r] == kse) ? (c[r] + bias) : NEG_;
            }
        }

        // ---- online softmax update (rows split across 16-lane halves) ----
        #pragma unroll
        for (int r = 0; r < 8; ++r) {
            float rowmax = sc[0][r];
            #pragma unroll
            for (int nt = 1; nt < 4; ++nt) rowmax = fmaxf(rowmax, sc[nt][r]);
            #pragma unroll
            for (int off = 1; off < 16; off <<= 1)
                rowmax = fmaxf(rowmax, __shfl_xor(rowmax, off, 16));

            const float mnew  = fmaxf(mrun[r], rowmax);
            const float alpha = __expf(mrun[r] - mnew);
            mrun[r] = mnew;

            float rs = 0.0f;
            #pragma unroll
            for (int nt = 0; nt < 4; ++nt) {
                const float p = __expf(sc[nt][r] - mnew);
                sc[nt][r] = p;
                rs += p;
            }
            #pragma unroll
            for (int off = 1; off < 16; off <<= 1)
                rs += __shfl_xor(rs, off, 16);

            lrun[r] = lrun[r] * alpha + rs;
            #pragma unroll
            for (int nt = 0; nt < 4; ++nt) accO[nt][r] *= alpha;
        }

        // ---- transpose P (C layout) -> A layout through per-wave LDS ----
        #pragma unroll
        for (int nt = 0; nt < 4; ++nt) {
            #pragma unroll
            for (int r = 0; r < 8; ++r)
                sP[wv][(r + 8 * half) * BN_ + nt * 16 + lx] = (__bf16)sc[nt][r];
        }
        asm volatile("s_wait_dscnt 0" ::: "memory");   // same-wave cross-lane exchange

        // A-frag K pattern = two contiguous 8-element runs per 32-chunk
        const __bf16* prow = &sP[wv][lx * BN_];
        v16bf pa0, pa1;
        {
            v8bf p0lo = *(const v8bf*)(prow + 8 * half);
            v8bf p0hi = *(const v8bf*)(prow + 16 + 8 * half);
            v8bf p1lo = *(const v8bf*)(prow + 32 + 8 * half);
            v8bf p1hi = *(const v8bf*)(prow + 48 + 8 * half);
            pa0 = __builtin_shufflevector(p0lo, p0hi, 0,1,2,3,4,5,6,7,8,9,10,11,12,13,14,15);
            pa1 = __builtin_shufflevector(p1lo, p1hi, 0,1,2,3,4,5,6,7,8,9,10,11,12,13,14,15);
        }

        // ---- O += P V : B-frag = (d=lx, keys in two 8-element runs like A) ----
        #pragma unroll
        for (int nt = 0; nt < 4; ++nt) {
            const __bf16* vrow = &sV[(nt * 16 + lx) * BN_];
            v16bf bv0 = __builtin_shufflevector(*(const v8bf*)(vrow + 8 * half), *(const v8bf*)(vrow + 16 + 8 * half), 0,1,2,3,4,5,6,7,8,9,10,11,12,13,14,15);
            v16bf bv1 = __builtin_shufflevector(*(const v8bf*)(vrow + 32 + 8 * half), *(const v8bf*)(vrow + 48 + 8 * half), 0,1,2,3,4,5,6,7,8,9,10,11,12,13,14,15);
            accO[nt] = __builtin_amdgcn_wmma_f32_16x16x32_bf16(false, pa0, false, bv0, (short)0, accO[nt], false, false);
            accO[nt] = __builtin_amdgcn_wmma_f32_16x16x32_bf16(false, pa1, false, bv1, (short)0, accO[nt], false, false);
        }
    }

    // ---- normalize and write out (f32) ----
    #pragma unroll
    for (int r = 0; r < 8; ++r) {
        float denom = lrun[r];
        denom = (denom == 0.0f) ? 1.0f : denom;
        const float inv = 1.0f / denom;
        const int m = qbase + r + 8 * half;
        float* op = out + (((size_t)b * H_ + h) * S_ + m) * D_;
        #pragma unroll
        for (int nt = 0; nt < 4; ++nt)
            op[nt * 16 + lx] = accO[nt][r] * inv;
    }
}

extern "C" void kernel_launch(void* const* d_in, const int* in_sizes, int n_in,
                              void* d_out, int out_size, void* d_ws, size_t ws_size,
                              hipStream_t stream) {
    const float* q      = (const float*)d_in[0];
    const float* k      = (const float*)d_in[1];
    const float* v      = (const float*)d_in[2];
    const int*   seg    = (const int*)  d_in[3];
    const int*   bidx   = (const int*)  d_in[4];
    const float* slopes = (const float*)d_in[5];
    float* out = (float*)d_out;

    dim3 grid(B_ * H_ * NB_);   // one workgroup per (b, h, query block)
    dim3 block(128);            // 4 wave32, one 16-row M-tile each
    hipLaunchKernelGGL(sparse_attn_wmma, grid, block, 0, stream,
                       q, k, v, seg, bidx, slopes, out);
}
